// TopDownParser_73400991088927
// MI455X (gfx1250) — matrix operand order repaired and checked by reading, not compiled, not run on hardware
//
#include <hip/hip_runtime.h>

// ---------------------------------------------------------------------------
// Types for CDNA5 WMMA (wave32): v_wmma_f32_16x16x32_bf16
// ---------------------------------------------------------------------------
typedef __attribute__((ext_vector_type(16))) __bf16 v16bf;
typedef __attribute__((ext_vector_type(8)))  __bf16 v8bf;
typedef __attribute__((ext_vector_type(8)))  float  v8f;

#define T_LEN 258     // N + 2
#define NP    257     // N + 1
#define DH    512     // D
#define GATES 2048    // 4*D
#define IN0   512     // TAG_D + WORD_D
#define IN1   1024    // 2*D
#define OUTC  129     // LABEL_V + 1

// Compiler scheduling fence: memory ops cannot be moved across this point.
#define SCHED_FENCE() asm volatile("" ::: "memory")

// Async global->LDS (gfx1250). Prototype (from hipcc diagnostic):
//   param0: int __vector(4) * (global src), param1: LDS dst, imm offset, imm cpol
#if defined(__HIP_DEVICE_COMPILE__) && __has_builtin(__builtin_amdgcn_global_load_async_to_lds_b128)
#define ASYNC_LDS 1
typedef int v4i_t __attribute__((vector_size(16)));
typedef v4i_t __attribute__((address_space(1)))* g1_v4i;
typedef v4i_t __attribute__((address_space(3)))* l3_v4i;
#else
#define ASYNC_LDS 0
#endif

__device__ __forceinline__ float sigf(float x) { return 1.f / (1.f + expf(-x)); }

// ---------------------------------------------------------------------------
// WMMA fragment loaders (ISA 7.12.2 layouts, wave32)
// A 16x32 bf16: lanes 0-15 row M=lane, K = {0..7, 16..23}; lanes 16-31 same
//               row group, K = {8..15, 24..31}
// B 32x16 bf16: lane n = column; lanes 0-15 hold K={0..7, 16..23}, lanes 16-31 K={8..15, 24..31}
// ---------------------------------------------------------------------------
__device__ __forceinline__ v16bf frag_a(const __bf16* __restrict__ row, int kk, int lane) {
    const __bf16* p = row + kk + ((lane & 16) ? 8 : 0);
    v8bf lo = *(const v8bf*)(p);
    v8bf hi = *(const v8bf*)(p + 16);
    v16bf r;
#pragma unroll
    for (int i = 0; i < 8; ++i) { r[i] = lo[i]; r[i + 8] = hi[i]; }
    return r;
}

__device__ __forceinline__ v16bf frag_b(const __bf16* __restrict__ Wt, int ldb, int n, int kk, int lane) {
    const __bf16* p = Wt + (size_t)n * ldb + kk + ((lane & 16) ? 8 : 0);
    v8bf lo = *(const v8bf*)(p);
    v8bf hi = *(const v8bf*)(p + 16);
    v16bf r;
#pragma unroll
    for (int i = 0; i < 8; ++i) { r[i] = lo[i]; r[i + 8] = hi[i]; }
    return r;
}

// Load the 4 B fragments of a 64-wide N strip for K-chunk kk.
__device__ __forceinline__ void loadB4(v16bf b[4], const __bf16* __restrict__ Wt,
                                       int ldb, int n0, int kk, int lane) {
#pragma unroll
    for (int nt = 0; nt < 4; ++nt)
        b[nt] = frag_b(Wt, ldb, n0 + nt * 16 + (lane & 15), kk, lane);
}

// ---------------------------------------------------------------------------
// Small prep kernels
// ---------------------------------------------------------------------------
__global__ void k_zero_ctr(unsigned* c) {
    if (threadIdx.x < 4) c[threadIdx.x] = 0u;
}

__global__ void k_f32_to_bf16(const float* __restrict__ s, __bf16* __restrict__ d, int n) {
    int i = blockIdx.x * 256 + threadIdx.x;
    if (i < n) d[i] = (__bf16)s[i];
}

// emb[t, 0:128] = tag_emb[tag_idx[t]], emb[t, 128:512] = word_emb[word_idx[t]]
__global__ void k_emb_gather(const int* __restrict__ tag_idx, const int* __restrict__ word_idx,
                             const float* __restrict__ tag_emb, const float* __restrict__ word_emb,
                             __bf16* __restrict__ emb) {
    int i = blockIdx.x * 256 + threadIdx.x;
    if (i >= T_LEN * IN0) return;
    int t = i >> 9, k = i & 511;
    float v = (k < 128) ? tag_emb[(size_t)tag_idx[t] * 128 + k]
                        : word_emb[(size_t)word_idx[t] * 384 + (k - 128)];
    emb[i] = (__bf16)v;
}

// WtLS[n, k] = (n<1024 ? Wl1[k, n] : Ws1[k, n-1024])   (2048 x 1024, bf16)
__global__ void k_build_wtls(const float* __restrict__ Wl1, const float* __restrict__ Ws1,
                             __bf16* __restrict__ Wt) {
    int i = blockIdx.x * 256 + threadIdx.x;
    if (i >= 2048 * 1024) return;
    int n = i >> 10, k = i & 1023;
    float v = (n < 1024) ? Wl1[(size_t)k * 1024 + n] : Ws1[(size_t)k * 1024 + (n - 1024)];
    Wt[i] = (__bf16)v;
}

// Wt2[n, k] = Wl2[k, n]   (128 x 1024, bf16)
__global__ void k_build_wtl2(const float* __restrict__ Wl2, __bf16* __restrict__ Wt) {
    int i = blockIdx.x * 256 + threadIdx.x;
    if (i >= 128 * 1024) return;
    int n = i >> 10, k = i & 1023;
    Wt[i] = (__bf16)Wl2[(size_t)k * 128 + n];
}

// X[x, 0:512] = h1[x, 0:512] ; X[x, 512:1024] = -h1[x+1, 512:1024]
__global__ void k_build_x(const float* __restrict__ h1, __bf16* __restrict__ X) {
    int i = blockIdx.x * 256 + threadIdx.x;
    if (i >= NP * 1024) return;
    int x = i >> 10, k = i & 1023;
    float v = (k < 512) ? h1[(size_t)x * 1024 + k] : -h1[(size_t)(x + 1) * 1024 + k];
    X[i] = (__bf16)v;
}

// ---------------------------------------------------------------------------
// Generic bf16 WMMA GEMM: D[z][m,n] = sum_k A[z][m,k]*Wt[z][n,k] + bias[z][n]
// A (M,K) bf16 row-major; Wt (N,K) bf16 row-major; D (M,N) fp32.
// Workgroup = 8 waves = (4 M-blocks x 2 N-halves) = 64M x 128N tile.
// Per wave: 16M x 64N (4 accum tiles), K step 32, explicit register
// double-buffering (ping-pong) so each WMMA consumes fragments loaded one
// full K-chunk earlier -> partial loadcnt waits, loads overlap the MAC chain.
// K must be a multiple of 64.
// ---------------------------------------------------------------------------
__global__ __launch_bounds__(256, 1)
void k_gemm_bf16(const __bf16* __restrict__ A, long long aStrZ,
                 const __bf16* __restrict__ Wt, long long wStrZ,
                 const float* __restrict__ bias, long long bStrZ,
                 float* __restrict__ D, long long dStrZ,
                 int M, int K, int N) {
    int lane = threadIdx.x & 31;
    int wave = threadIdx.x >> 5;
    int mb = wave & 3, nh = wave >> 2;
    int z = blockIdx.z;
    A  += (size_t)z * aStrZ;
    Wt += (size_t)z * wStrZ;
    D  += (size_t)z * dStrZ;
    if (bias) bias += (size_t)z * bStrZ;

    int m0 = blockIdx.x * 64 + mb * 16;
    int n0 = blockIdx.y * 128 + nh * 64;
    int mrow = m0 + (lane & 15);
    if (mrow >= M) mrow = M - 1;
    const __bf16* arow = A + (size_t)mrow * K;

    v8f acc[4];
#pragma unroll
    for (int nt = 0; nt < 4; ++nt) acc[nt] = 0.f;

    v16bf a0 = frag_a(arow, 0, lane), a1;
    v16bf b0[4], b1[4];
    loadB4(b0, Wt, K, n0, 0, lane);

    for (int kk = 0; kk < K; kk += 64) {
        // prefetch chunk kk+32
        a1 = frag_a(arow, kk + 32, lane);
        loadB4(b1, Wt, K, n0, kk + 32, lane);
        SCHED_FENCE();
#pragma unroll
        for (int nt = 0; nt < 4; ++nt)
            acc[nt] = __builtin_amdgcn_wmma_f32_16x16x32_bf16(
                false, a0, false, b0[nt], (short)0, acc[nt], false, false);
        // prefetch chunk kk+64 (clamped: last iteration loads chunk 0, unused)
        int kk2 = (kk + 64 < K) ? (kk + 64) : 0;
        a0 = frag_a(arow, kk2, lane);
        loadB4(b0, Wt, K, n0, kk2, lane);
        SCHED_FENCE();
#pragma unroll
        for (int nt = 0; nt < 4; ++nt)
            acc[nt] = __builtin_amdgcn_wmma_f32_16x16x32_bf16(
                false, a1, false, b1[nt], (short)0, acc[nt], false, false);
    }

    int rbase = m0 + ((lane & 16) ? 8 : 0);
#pragma unroll
    for (int nt = 0; nt < 4; ++nt) {
        int col = n0 + nt * 16 + (lane & 15);
        float bv = bias ? bias[col] : 0.f;
#pragma unroll
        for (int v = 0; v < 8; ++v) {
            int rr = rbase + v;
            if (rr < M) D[(size_t)rr * N + col] = acc[nt][v] + bv;
        }
    }
}

// ---------------------------------------------------------------------------
// BiLSTM recurrence. Grid (16, 2): 16 WGs per direction, 256 threads each.
// Each WG owns h-slice j in [wg*32, wg*32+32): keeps the 128 Whh rows
// {j, 512+j, 1024+j, 1536+j} resident in LDS (fp32, stride 513 -> conflict
// free banks). Whh preload uses gfx1250 async global->LDS DMA.
// Per step: 128 dots of 512 across 256 threads, gate nonlinearity on 32
// threads, h slice to global, then device-scope atomic spin barrier.
// LDS: 128*513 (Whh) + 512 (h) + 256 (partial) + 128 (g) + 32 (c) floats.
// ---------------------------------------------------------------------------
#define REC_LDS_FLOATS (128 * 513 + 512 + 256 + 128 + 32)

__global__ __launch_bounds__(256)
void k_lstm_rec(const float* __restrict__ Whh,   // (2, 2048, 512)
                const float* __restrict__ pre,   // (2, 258, 2048)
                float* __restrict__ hs,          // (258, 1024), dir offset dir*512
                __bf16* __restrict__ hs_bf,      // same, bf16
                unsigned* __restrict__ counters) // 2 counters (one per dir)
{
    extern __shared__ float lds[];
    float* w_lds    = lds;                       // 128 * 513
    float* h_lds    = w_lds + 128 * 513;         // 512
    float* partial  = h_lds + 512;               // 256
    float* g_lds    = partial + 256;             // 128
    float* c_lds    = g_lds + 128;               // 32

    const int tid  = threadIdx.x;
    const int wg   = blockIdx.x;   // 0..15
    const int dir  = blockIdx.y;   // 0..1
    const int base = wg * 32;

    const float* WhhD = Whh + (size_t)dir * GATES * DH;
#if ASYNC_LDS
    {
        // 128 rows x 128 16-byte chunks, async DMA-style copies to LDS.
        for (int i = tid; i < 128 * (DH / 4); i += 256) {
            int row = i >> 7, kc = i & 127;              // kc: 4-float chunk
            int q = row >> 5, jj = row & 31;
            const float* g = WhhD + (size_t)(q * 512 + base + jj) * DH + kc * 4;
            float* l = &w_lds[row * 513 + kc * 4];
            __builtin_amdgcn_global_load_async_to_lds_b128(
                (g1_v4i)g, (l3_v4i)l, 0, 0);
        }
#if __has_builtin(__builtin_amdgcn_s_wait_asynccnt)
        __builtin_amdgcn_s_wait_asynccnt(0);
#else
        asm volatile("s_wait_asynccnt 0x0" ::: "memory");
#endif
    }
#else
    for (int i = tid; i < 128 * DH; i += 256) {
        int row = i / DH, k = i % DH;                    // row = q*32 + jj
        int q = row >> 5, jj = row & 31;
        w_lds[row * 513 + k] = WhhD[(size_t)(q * 512 + base + jj) * DH + k];
    }
#endif
    if (tid < 32) c_lds[tid] = 0.f;
    __syncthreads();

    const int row  = tid & 127;
    const int half = tid >> 7;
    unsigned* ctr = counters + dir;

    for (int s = 0; s < T_LEN; ++s) {
        int t = dir ? (T_LEN - 1 - s) : s;
        // previous h -> LDS
        if (s == 0) {
            for (int i = tid; i < 512; i += 256) h_lds[i] = 0.f;
        } else {
            int tp = dir ? (t + 1) : (t - 1);
            for (int i = tid; i < 512; i += 256)
                h_lds[i] = hs[(size_t)tp * 1024 + dir * 512 + i];
        }
        __syncthreads();

        // 128 dots of 512, split in halves across 256 threads
        float acc = 0.f;
        const float* wr = &w_lds[row * 513 + half * 256];
        const float* hr = &h_lds[half * 256];
#pragma unroll 8
        for (int k = 0; k < 256; ++k) acc += wr[k] * hr[k];
        partial[tid] = acc;
        __syncthreads();

        if (tid < 128) {
            int q = tid >> 5, jj = tid & 31;
            g_lds[tid] = partial[tid] + partial[tid + 128]
                       + pre[((size_t)dir * T_LEN + t) * GATES + q * 512 + base + jj];
        }
        __syncthreads();

        if (tid < 32) {
            float gi = g_lds[tid], gf = g_lds[32 + tid];
            float gg = g_lds[64 + tid], go = g_lds[96 + tid];
            float c = sigf(gf) * c_lds[tid] + sigf(gi) * tanhf(gg);
            float h = sigf(go) * tanhf(c);
            c_lds[tid] = c;
            size_t oi = (size_t)t * 1024 + dir * 512 + base + tid;
            hs[oi]    = h;
            hs_bf[oi] = (__bf16)h;
        }

        if (s < T_LEN - 1) {
            __threadfence();      // release h stores to device scope
            __syncthreads();
            if (tid == 0) {
                __hip_atomic_fetch_add(ctr, 1u, __ATOMIC_RELEASE, __HIP_MEMORY_SCOPE_AGENT);
                unsigned target = 16u * (unsigned)(s + 1);
                while (__hip_atomic_load(ctr, __ATOMIC_ACQUIRE, __HIP_MEMORY_SCOPE_AGENT) < target)
                    __builtin_amdgcn_s_sleep(1);
            }
            __syncthreads();
            __threadfence();      // acquire: see other WGs' h stores
        }
    }
}

// ---------------------------------------------------------------------------
// Span label head: out[l,r,0:128] = relu(PL[r]-PL[l]+bl1) @ Wl2 + bl2,
// masked to r>l else 0. A fragments generated on the fly (fp32 -> relu ->
// bf16); same register double-buffered pipeline as k_gemm_bf16.
// Per wave: 16 r x 64 cols; WG: 64 r x 128 cols. Grid: (5, 257).
// PQ: (257, 2048) fp32, PL = cols 0..1023.
// ---------------------------------------------------------------------------
__device__ __forceinline__ v16bf build_a(const float* __restrict__ plr,
                                         const float* __restrict__ pll,
                                         const float* __restrict__ bl1,
                                         int kk, int lane) {
    int ka = kk + ((lane & 16) ? 8 : 0);
    v16bf a;
#pragma unroll
    for (int i = 0; i < 8; ++i) {
        float x1 = plr[ka + i]      - pll[ka + i]      + bl1[ka + i];
        float x2 = plr[ka + 16 + i] - pll[ka + 16 + i] + bl1[ka + 16 + i];
        a[i]     = (__bf16)fmaxf(x1, 0.f);
        a[i + 8] = (__bf16)fmaxf(x2, 0.f);
    }
    return a;
}

__global__ __launch_bounds__(256, 1)
void k_span_label(const float* __restrict__ PQ, const __bf16* __restrict__ Wt2,
                  const float* __restrict__ bl1, const float* __restrict__ bl2,
                  float* __restrict__ out) {
    int lane = threadIdx.x & 31;
    int wave = threadIdx.x >> 5;
    int mb = wave & 3, nh = wave >> 2;
    int l  = blockIdx.y;
    int r0 = blockIdx.x * 64 + mb * 16;
    int n0 = nh * 64;
    int rr = r0 + (lane & 15);
    if (rr >= NP) rr = NP - 1;
    const float* plr = PQ + (size_t)rr * 2048;
    const float* pll = PQ + (size_t)l  * 2048;

    v8f acc[4];
#pragma unroll
    for (int nt = 0; nt < 4; ++nt) acc[nt] = 0.f;

    v16bf a0 = build_a(plr, pll, bl1, 0, lane), a1;
    v16bf b0[4], b1[4];
    loadB4(b0, Wt2, 1024, n0, 0, lane);

    for (int kk = 0; kk < 1024; kk += 64) {
        a1 = build_a(plr, pll, bl1, kk + 32, lane);
        loadB4(b1, Wt2, 1024, n0, kk + 32, lane);
        SCHED_FENCE();
#pragma unroll
        for (int nt = 0; nt < 4; ++nt)
            acc[nt] = __builtin_amdgcn_wmma_f32_16x16x32_bf16(
                false, a0, false, b0[nt], (short)0, acc[nt], false, false);
        int kk2 = (kk + 64 < 1024) ? (kk + 64) : 0;
        a0 = build_a(plr, pll, bl1, kk2, lane);
        loadB4(b0, Wt2, 1024, n0, kk2, lane);
        SCHED_FENCE();
#pragma unroll
        for (int nt = 0; nt < 4; ++nt)
            acc[nt] = __builtin_amdgcn_wmma_f32_16x16x32_bf16(
                false, a1, false, b1[nt], (short)0, acc[nt], false, false);
    }

    int rbase = r0 + ((lane & 16) ? 8 : 0);
#pragma unroll
    for (int nt = 0; nt < 4; ++nt) {
        int col = n0 + nt * 16 + (lane & 15);
        float bv = bl2[col];
#pragma unroll
        for (int v = 0; v < 8; ++v) {
            int r = rbase + v;
            if (r < NP)
                out[((size_t)l * NP + r) * OUTC + col] = (r > l) ? (acc[nt][v] + bv) : 0.f;
        }
    }
}

// ---------------------------------------------------------------------------
// Split head: out[l,r,128] = relu(QS[r]-QS[l]+bs1) . Ws2 + bs2 (masked).
// One wave per (l,r) pair, lanes over k, shuffle reduce.
// ---------------------------------------------------------------------------
__global__ __launch_bounds__(256)
void k_span_split(const float* __restrict__ PQ, const float* __restrict__ Ws2,
                  const float* __restrict__ bs1, const float* __restrict__ bs2,
                  float* __restrict__ out) {
    int lane = threadIdx.x & 31;
    int w = threadIdx.x >> 5;
    long long p = (long long)blockIdx.x * 8 + w;
    if (p >= (long long)NP * NP) return;
    int l = (int)(p / NP), r = (int)(p % NP);
    const float* qr = PQ + (size_t)r * 2048 + 1024;
    const float* ql = PQ + (size_t)l * 2048 + 1024;
    float acc = 0.f;
    for (int k = lane; k < 1024; k += 32) {
        float v = qr[k] - ql[k] + bs1[k];
        acc += fmaxf(v, 0.f) * Ws2[k];
    }
#pragma unroll
    for (int off = 16; off > 0; off >>= 1) acc += __shfl_xor(acc, off, 32);
    if (lane == 0) out[(size_t)p * OUTC + 128] = (r > l) ? (acc + bs2[0]) : 0.f;
}

// ---------------------------------------------------------------------------
// Host-side launch
// ---------------------------------------------------------------------------
extern "C" void kernel_launch(void* const* d_in, const int* in_sizes, int n_in,
                              void* d_out, int out_size, void* d_ws, size_t ws_size,
                              hipStream_t stream) {
    (void)in_sizes; (void)n_in; (void)out_size; (void)ws_size;

    const int*   tag_idx  = (const int*)d_in[0];
    const int*   word_idx = (const int*)d_in[1];
    const float* tag_emb  = (const float*)d_in[2];
    const float* word_emb = (const float*)d_in[3];
    const float* Wih0 = (const float*)d_in[4];
    const float* Whh0 = (const float*)d_in[5];
    const float* b0   = (const float*)d_in[6];
    const float* Wih1 = (const float*)d_in[7];
    const float* Whh1 = (const float*)d_in[8];
    const float* b1   = (const float*)d_in[9];
    const float* Wl1  = (const float*)d_in[10];
    const float* bl1  = (const float*)d_in[11];
    const float* Wl2  = (const float*)d_in[12];
    const float* bl2  = (const float*)d_in[13];
    const float* Ws1  = (const float*)d_in[14];
    const float* bs1  = (const float*)d_in[15];
    const float* Ws2  = (const float*)d_in[16];
    const float* bs2  = (const float*)d_in[17];
    float* out = (float*)d_out;

    // Workspace carve
    char* ws = (char*)d_ws;
    size_t off = 0;
    auto carve = [&](size_t bytes) -> void* {
        void* p = ws + off;
        off = (off + bytes + 255) & ~(size_t)255;
        return p;
    };
    __bf16* emb_bf  = (__bf16*)carve((size_t)T_LEN * IN0 * 2);
    __bf16* wih0_bf = (__bf16*)carve((size_t)2 * GATES * IN0 * 2);
    __bf16* wih1_bf = (__bf16*)carve((size_t)2 * GATES * IN1 * 2);
    __bf16* wtls_bf = (__bf16*)carve((size_t)2048 * 1024 * 2);
    __bf16* wtl2_bf = (__bf16*)carve((size_t)128 * 1024 * 2);
    float*  pre0    = (float*)carve((size_t)2 * T_LEN * GATES * 4);
    float*  pre1    = (float*)carve((size_t)2 * T_LEN * GATES * 4);
    float*  h0      = (float*)carve((size_t)T_LEN * 1024 * 4);
    __bf16* h0_bf   = (__bf16*)carve((size_t)T_LEN * 1024 * 2);
    float*  h1      = (float*)carve((size_t)T_LEN * 1024 * 4);
    __bf16* h1_bf   = (__bf16*)carve((size_t)T_LEN * 1024 * 2);
    __bf16* x_bf    = (__bf16*)carve((size_t)NP * 1024 * 2);
    float*  pq      = (float*)carve((size_t)NP * 2048 * 4);
    unsigned* ctrs  = (unsigned*)carve(4 * sizeof(unsigned));

    const size_t recLds = (size_t)REC_LDS_FLOATS * 4;

    // 0) zero spin-barrier counters (fresh every call: deterministic)
    k_zero_ctr<<<1, 32, 0, stream>>>(ctrs);

    // 1) weight / input prep (independent)
    k_f32_to_bf16<<<(2 * GATES * IN0 + 255) / 256, 256, 0, stream>>>(Wih0, wih0_bf, 2 * GATES * IN0);
    k_f32_to_bf16<<<(2 * GATES * IN1 + 255) / 256, 256, 0, stream>>>(Wih1, wih1_bf, 2 * GATES * IN1);
    k_build_wtls<<<(2048 * 1024 + 255) / 256, 256, 0, stream>>>(Wl1, Ws1, wtls_bf);
    k_build_wtl2<<<(128 * 1024 + 255) / 256, 256, 0, stream>>>(Wl2, wtl2_bf);
    k_emb_gather<<<(T_LEN * IN0 + 255) / 256, 256, 0, stream>>>(tag_idx, word_idx, tag_emb, word_emb, emb_bf);

    // 2) layer-0 input projection: pre0[d] = emb @ Wih0[d]^T + b0[d]
    k_gemm_bf16<<<dim3(5, GATES / 128, 2), 256, 0, stream>>>(
        emb_bf, 0, wih0_bf, (long long)GATES * IN0, b0, GATES,
        pre0, (long long)T_LEN * GATES, T_LEN, IN0, GATES);

    // 3) layer-0 recurrence (16 WGs x 2 dirs, LDS-resident Whh slices)
    k_lstm_rec<<<dim3(16, 2), 256, recLds, stream>>>(Whh0, pre0, h0, h0_bf, ctrs);

    // 4) layer-1 input projection: pre1[d] = h0 @ Wih1[d]^T + b1[d]
    k_gemm_bf16<<<dim3(5, GATES / 128, 2), 256, 0, stream>>>(
        h0_bf, 0, wih1_bf, (long long)GATES * IN1, b1, GATES,
        pre1, (long long)T_LEN * GATES, T_LEN, IN1, GATES);

    // 5) layer-1 recurrence
    k_lstm_rec<<<dim3(16, 2), 256, recLds, stream>>>(Whh1, pre1, h1, h1_bf, ctrs + 2);

    // 6) X = [Fp | -Bp] in bf16
    k_build_x<<<(NP * 1024 + 255) / 256, 256, 0, stream>>>(h1, x_bf);

    // 7) PQ = X @ [Wl1 | Ws1]  (257 x 2048)
    k_gemm_bf16<<<dim3(5, 2048 / 128, 1), 256, 0, stream>>>(
        x_bf, 0, wtls_bf, 0, (const float*)nullptr, 0,
        pq, 0, NP, 1024, 2048);

    // 8) label head (dominant WMMA GEMM) + split head
    k_span_label<<<dim3(5, NP), 256, 0, stream>>>(pq, wtl2_bf, bl1, bl2, out);
    k_span_split<<<((NP * NP) + 7) / 8, 256, 0, stream>>>(pq, Ws2, bs1, bs2, out);
}
